// MoEMLPBlock_558345749170
// MI455X (gfx1250) — matrix operand order repaired and checked by reading, not compiled, not run on hardware
//
#include <hip/hip_runtime.h>
#include <hip/hip_bf16.h>
#include <math.h>

// ---------------- problem constants (match reference) ----------------
#define T_TOK 4096   // B*S tokens
#define D_DIM 1024   // embedding
#define H_DIM 4096   // mlp dim
#define E_NUM 64     // experts
#define K_TOP 3
#define C_CAP 512    // expert capacity
#define ECHUNK 8     // experts per GEMM launch (h buffer = ECHUNK*C*H bf16 = 32MB)

typedef __attribute__((ext_vector_type(16))) __bf16 v16bf;
typedef __attribute__((ext_vector_type(8)))  float  v8f;

// ---------------- workspace layout (bytes) ----------------
#define OFF_TI      0u            // int   [T*K]      topk expert ids
#define OFF_TG      65536u        // float [T*K]      topk gates
#define OFF_INVTOK  131072u       // int   [E*C]      slot -> token (-1 empty)
#define OFF_INVG    262144u       // float [E*C]      slot -> gate
#define OFF_BUF     393216u       // bf16  [E*C*D]    dispatched tokens (64MB)
#define OFF_H       67502080u     // bf16  [ECHUNK*C*H] hidden (32MB)

// =====================================================================
// Gating: logits = x @ gate_w + gate_b ; top-3 ; softmax over top-3
// =====================================================================
__global__ __launch_bounds__(64) void moe_gate_k(const float* __restrict__ x,
                                                 const float* __restrict__ gw,
                                                 const float* __restrict__ gb,
                                                 int* __restrict__ ti,
                                                 float* __restrict__ tg) {
  __shared__ float xs[D_DIM];
  __shared__ float lg[E_NUM];
  const int t = blockIdx.x, tid = threadIdx.x;
#pragma unroll
  for (int i = 0; i < D_DIM / 64; ++i) xs[tid + i * 64] = x[(size_t)t * D_DIM + tid + i * 64];
  __syncthreads();
  float acc = gb[tid];
#pragma unroll 8
  for (int d = 0; d < D_DIM; ++d) acc += xs[d] * gw[d * E_NUM + tid];
  lg[tid] = acc;
  __syncthreads();
  if (tid == 0) {
    int   idx[K_TOP];
    float val[K_TOP];
    for (int k = 0; k < K_TOP; ++k) {
      float best = -3.0e38f; int bi = 0;
      for (int e = 0; e < E_NUM; ++e)
        if (lg[e] > best) { best = lg[e]; bi = e; }   // first occurrence wins (matches top_k)
      idx[k] = bi; val[k] = best; lg[bi] = -3.0e38f;
    }
    const float m = val[0];
    float ex[K_TOP], s = 0.f;
    for (int k = 0; k < K_TOP; ++k) { ex[k] = expf(val[k] - m); s += ex[k]; }
    for (int k = 0; k < K_TOP; ++k) { ti[t * K_TOP + k] = idx[k]; tg[t * K_TOP + k] = ex[k] / s; }
  }
}

// =====================================================================
// Dispatch: order-preserving per-expert cumsum over the flat [T*K] list.
// =====================================================================
__global__ __launch_bounds__(64) void moe_dispatch_k(const int* __restrict__ ti,
                                                     const float* __restrict__ tg,
                                                     int* __restrict__ inv_tok,
                                                     float* __restrict__ inv_g) {
  const int e = threadIdx.x;
  int cnt = 0;
  for (int a = 0; a < T_TOK * K_TOP; ++a) {
    if (ti[a] == e) {
      if (cnt < C_CAP) { inv_tok[e * C_CAP + cnt] = a / K_TOP; inv_g[e * C_CAP + cnt] = tg[a]; }
      ++cnt;
    }
  }
  for (int c = (cnt < C_CAP ? cnt : C_CAP); c < C_CAP; ++c) inv_tok[e * C_CAP + c] = -1;
}

// =====================================================================
// Scatter: copy token row (fp32) into expert buffer (bf16).
// =====================================================================
__global__ __launch_bounds__(256) void moe_scatter_k(const float* __restrict__ x,
                                                     const int* __restrict__ inv_tok,
                                                     __bf16* __restrict__ buf) {
  const int slot = blockIdx.x;           // e*C + c
  const int tok = inv_tok[slot];
  if (tok < 0) return;
  const int tid = threadIdx.x;
#pragma unroll
  for (int j = 0; j < D_DIM / 256; ++j) {
    const int cc = tid + j * 256;
    buf[(size_t)slot * D_DIM + cc] = (__bf16)x[(size_t)tok * D_DIM + cc];
  }
}

// =====================================================================
// WMMA tile engine: 256 threads / 8 waves, 128x128 tile, K-stage 64.
//  * A (bf16, no conversion): async global->LDS DMA, double-buffered LDS,
//    tracked with ASYNCcnt (s_wait_asynccnt); zero VGPR staging cost.
//  * B (fp32 weights): register-staged pipeline, v_cvt_pk_bf16_f32 packed
//    2-per-dword ds_store_b32, stored transposed [n][k].
// Per stage: barrier ; commitB ; async A(k+64) ; wait async<=4 ; barrier ;
//            stage B(k+64) ; 2 K-slices x 8 WMMAs.
// LDS rows padded to 72 elems (144B): 16B-aligned, bank-conflict-free.
// =====================================================================
#define LSTR 72  // padded LDS row stride in bf16 elements (144 bytes)

__device__ __forceinline__ v16bf ld_frag(const __bf16* p0, const __bf16* p1) {
  union { uint4 q[2]; v16bf v; } u;
  u.q[0] = *(const uint4*)p0;
  u.q[1] = *(const uint4*)p1;
  return u.v;
}

__device__ __forceinline__ unsigned pack_bf16(float a, float b) {
#if __has_builtin(__builtin_amdgcn_cvt_pk_bf16_f32)
  auto v = __builtin_amdgcn_cvt_pk_bf16_f32(a, b);   // single v_cvt_pk_bf16_f32
  return __builtin_bit_cast(unsigned, v);
#else
  union { __bf16 h[2]; unsigned u; } u;
  u.h[0] = (__bf16)a;
  u.h[1] = (__bf16)b;
  return u.u;
#endif
}

__device__ __forceinline__ void async_cp16(const __bf16* gsrc, __bf16* ldst) {
  const unsigned loff = (unsigned)(size_t)ldst;                 // LDS byte offset (low 32 bits)
  const unsigned long long ga = (unsigned long long)(size_t)gsrc;
  asm volatile("global_load_async_to_lds_b128 %0, %1, off"
               :: "v"(loff), "v"(ga) : "memory");
}

__device__ __forceinline__ void wait_async_le4() {
#if __has_builtin(__builtin_amdgcn_s_wait_asynccnt)
  __builtin_amdgcn_s_wait_asynccnt(4);
#else
  asm volatile("s_wait_asynccnt 0x4" ::: "memory");
#endif
}

__device__ __forceinline__ void wait_async_le0() {
#if __has_builtin(__builtin_amdgcn_s_wait_asynccnt)
  __builtin_amdgcn_s_wait_asynccnt(0);
#else
  asm volatile("s_wait_asynccnt 0x0" ::: "memory");
#endif
}

__device__ __forceinline__ void mma_loop(const __bf16* __restrict__ Ag, int lda,
                                         const float* __restrict__ Bg, int ldb,
                                         int nK, __bf16* Alds0, __bf16* Alds1,
                                         __bf16* Blds, v8f (&acc)[4][2]) {
  const int tid = threadIdx.x;
  const int lane = tid & 31, wave = tid >> 5;
  const int wm = wave >> 2, wn = wave & 3;
  const int half = lane >> 4, l16 = lane & 15;

  // B staged-tile registers: 64x128 fp32 as 4 k-pair float4 couples/thread
  float4 pb0[4], pb1[4];

  auto asyncA = [&](int kt, __bf16* dst) {   // 4 x 16B per thread, DMA to LDS
#pragma unroll
    for (int i = 0; i < 4; ++i) {
      const int lin = tid + i * 256;
      const int row = lin >> 3, c8 = (lin & 7) * 8;
      async_cp16(&Ag[(size_t)row * lda + kt + c8], &dst[row * LSTR + c8]);
    }
  };

  auto stageB = [&](int kt) {
    const float* Bgk = Bg + (size_t)kt * ldb;
#pragma unroll
    for (int i = 0; i < 4; ++i) {
      const int p = tid + i * 256;
      const int k = (p >> 5) * 2, n4 = (p & 31) * 4;
      pb0[i] = *(const float4*)&Bgk[(size_t)k * ldb + n4];
      pb1[i] = *(const float4*)&Bgk[(size_t)(k + 1) * ldb + n4];
    }
  };

  auto commitB = [&]() {
#pragma unroll
    for (int i = 0; i < 4; ++i) {
      const int p = tid + i * 256;
      const int k = (p >> 5) * 2, n4 = (p & 31) * 4;
      const float4 lo = pb0[i], hi = pb1[i];
      *(unsigned*)&Blds[(n4 + 0) * LSTR + k] = pack_bf16(lo.x, hi.x);
      *(unsigned*)&Blds[(n4 + 1) * LSTR + k] = pack_bf16(lo.y, hi.y);
      *(unsigned*)&Blds[(n4 + 2) * LSTR + k] = pack_bf16(lo.z, hi.z);
      *(unsigned*)&Blds[(n4 + 3) * LSTR + k] = pack_bf16(lo.w, hi.w);
    }
  };

  auto compute = [&](const __bf16* Acur) {
#pragma unroll
    for (int s = 0; s < 2; ++s) {
      v16bf af[4], bfr[2];
#pragma unroll
      for (int f = 0; f < 4; ++f) { // A frag: row M=l16, quads K = s*32 + {half*8, 16+half*8}
        const __bf16* base = &Acur[(wm * 64 + f * 16 + l16) * LSTR + s * 32];
        af[f] = ld_frag(base + half * 8, base + 16 + half * 8);
      }
#pragma unroll
      for (int g = 0; g < 2; ++g) { // B frag: col N=l16, quads K = s*32 + {half*8, 16+half*8}
        const __bf16* base = &Blds[(wn * 32 + g * 16 + l16) * LSTR + s * 32];
        bfr[g] = ld_frag(base + half * 8, base + 16 + half * 8);
      }
#pragma unroll
      for (int f = 0; f < 4; ++f)
#pragma unroll
        for (int g = 0; g < 2; ++g)
          acc[f][g] = __builtin_amdgcn_wmma_f32_16x16x32_bf16(
              false, af[f], false, bfr[g], (short)0, acc[f][g], false, false);
    }
  };

  // prologue: DMA A tile 0, stage B tile 0
  asyncA(0, Alds0);
  stageB(0);
  __bf16* Acur = Alds0;
  __bf16* Anxt = Alds1;
  int kt = 0;
  for (; kt + 64 < nK; kt += 64) {
    __syncthreads();        // all waves done reading previous LDS tiles
    commitB();
    asyncA(kt + 64, Anxt);  // DMA next A tile; lands during compute
    wait_async_le4();       // A(kt) batch (issued last stage) has landed
    __syncthreads();        // publish A(kt) + B(kt) to all waves
    stageB(kt + 64);        // next B global loads, hidden by WMMAs
    if (kt + 128 < nK) {    // L2 pre-warm two stages ahead on fp32 weight stream
      const float* pf = &Bg[(size_t)(kt + 128 + (tid >> 2)) * ldb + (tid & 3) * 32];
      __builtin_prefetch(pf, 0, 1);
    }
    compute(Acur);
    __bf16* t = Acur; Acur = Anxt; Anxt = t;
  }
  // peeled final stage
  __syncthreads();
  commitB();
  wait_async_le0();
  __syncthreads();
  compute(Acur);
}

// =====================================================================
// GEMM1: h = GELU(buf @ w1 + b1)   (per expert, bf16 out)
// =====================================================================
__global__ __launch_bounds__(256, 1) void moe_gemm1_k(const __bf16* __restrict__ buf,
                                                      const float* __restrict__ w1,
                                                      const float* __restrict__ b1,
                                                      __bf16* __restrict__ h, int chunk) {
  __shared__ __bf16 AldsA[128 * LSTR];
  __shared__ __bf16 AldsB[128 * LSTR];
  __shared__ __bf16 Blds[128 * LSTR];
  const int mt = blockIdx.x & 3, nt = blockIdx.x >> 2;
  const int el = blockIdx.y, eg = chunk * ECHUNK + el;
  const __bf16* Ag = buf + ((size_t)eg * C_CAP + mt * 128) * D_DIM;
  const float*  Bg = w1 + (size_t)eg * D_DIM * H_DIM + nt * 128;
  v8f acc[4][2] = {};
  mma_loop(Ag, D_DIM, Bg, H_DIM, D_DIM, AldsA, AldsB, Blds, acc);

  const int lane = threadIdx.x & 31, wave = threadIdx.x >> 5;
  const int wm = wave >> 2, wn = wave & 3, half = lane >> 4, l16 = lane & 15;
#pragma unroll
  for (int f = 0; f < 4; ++f)
#pragma unroll
    for (int g = 0; g < 2; ++g) {
      const int nG = nt * 128 + wn * 32 + g * 16 + l16;
      const float bb = b1[(size_t)eg * H_DIM + nG];
#pragma unroll
      for (int r = 0; r < 8; ++r) {
        const int m = mt * 128 + wm * 64 + f * 16 + half * 8 + r;
        const float v = acc[f][g][r] + bb;
        const float ge = 0.5f * v * (1.0f + erff(v * 0.70710678118654752f)); // exact GELU
        h[((size_t)el * C_CAP + m) * H_DIM + nG] = (__bf16)ge;
      }
    }
}

// =====================================================================
// GEMM2 + fused combine: y = h @ w2 + b2 ; out[tok] += gate * y_row
// =====================================================================
__global__ __launch_bounds__(256, 1) void moe_gemm2_k(const __bf16* __restrict__ h,
                                                      const float* __restrict__ w2,
                                                      const float* __restrict__ b2,
                                                      const int* __restrict__ inv_tok,
                                                      const float* __restrict__ inv_g,
                                                      float* __restrict__ out, int chunk) {
  __shared__ __bf16 AldsA[128 * LSTR];
  __shared__ __bf16 AldsB[128 * LSTR];
  __shared__ __bf16 Blds[128 * LSTR];
  const int mt = blockIdx.x & 3, nt = blockIdx.x >> 2;
  const int el = blockIdx.y, eg = chunk * ECHUNK + el;
  const __bf16* Ag = h + ((size_t)el * C_CAP + mt * 128) * H_DIM;
  const float*  Bg = w2 + (size_t)eg * H_DIM * D_DIM + nt * 128;
  v8f acc[4][2] = {};
  mma_loop(Ag, H_DIM, Bg, D_DIM, H_DIM, AldsA, AldsB, Blds, acc);

  const int lane = threadIdx.x & 31, wave = threadIdx.x >> 5;
  const int wm = wave >> 2, wn = wave & 3, half = lane >> 4, l16 = lane & 15;
#pragma unroll
  for (int f = 0; f < 4; ++f)
#pragma unroll
    for (int r = 0; r < 8; ++r) {
      const int c = mt * 128 + wm * 64 + f * 16 + half * 8 + r;
      const int tok = inv_tok[eg * C_CAP + c];
      if (tok < 0) continue;                       // empty slot: never combined
      const float gate = inv_g[eg * C_CAP + c];
#pragma unroll
      for (int g = 0; g < 2; ++g) {
        const int nG = nt * 128 + wn * 32 + g * 16 + l16;
        const float val = (acc[f][g][r] + b2[(size_t)eg * D_DIM + nG]) * gate;
        __hip_atomic_fetch_add(&out[(size_t)tok * D_DIM + nG], val,
                               __ATOMIC_RELAXED, __HIP_MEMORY_SCOPE_AGENT);
      }
    }
}

// =====================================================================
extern "C" void kernel_launch(void* const* d_in, const int* in_sizes, int n_in,
                              void* d_out, int out_size, void* d_ws, size_t ws_size,
                              hipStream_t stream) {
  const float* x      = (const float*)d_in[0];
  const float* gate_w = (const float*)d_in[1];
  const float* gate_b = (const float*)d_in[2];
  const float* w1     = (const float*)d_in[3];
  const float* b1     = (const float*)d_in[4];
  const float* w2     = (const float*)d_in[5];
  const float* b2     = (const float*)d_in[6];
  float* out = (float*)d_out;

  char* ws = (char*)d_ws;
  int*    ti      = (int*)(ws + OFF_TI);
  float*  tg      = (float*)(ws + OFF_TG);
  int*    inv_tok = (int*)(ws + OFF_INVTOK);
  float*  inv_g   = (float*)(ws + OFF_INVG);
  __bf16* buf     = (__bf16*)(ws + OFF_BUF);
  __bf16* hbuf    = (__bf16*)(ws + OFF_H);

  hipMemsetAsync(d_out, 0, (size_t)T_TOK * D_DIM * sizeof(float), stream);

  moe_gate_k<<<T_TOK, 64, 0, stream>>>(x, gate_w, gate_b, ti, tg);
  moe_dispatch_k<<<1, 64, 0, stream>>>(ti, tg, inv_tok, inv_g);
  moe_scatter_k<<<E_NUM * C_CAP, 256, 0, stream>>>(x, inv_tok, buf);

  for (int chunk = 0; chunk < E_NUM / ECHUNK; ++chunk) {
    moe_gemm1_k<<<dim3(4 * (H_DIM / 128), ECHUNK), 256, 0, stream>>>(buf, w1, b1, hbuf, chunk);
    moe_gemm2_k<<<dim3(4 * (D_DIM / 128), ECHUNK), 256, 0, stream>>>(hbuf, w2, b2,
                                                                     inv_tok, inv_g, out, chunk);
  }
}
